// TabTransformer_85693187490014
// MI455X (gfx1250) — matrix operand order repaired and checked by reading, not compiled, not run on hardware
//
#include <hip/hip_runtime.h>
#include <hip/hip_bf16.h>
#include <math.h>

// ---------------------------------------------------------------------------
// TabTransformer forward, CDNA5 (gfx1250, wave32).
// - All dense GEMMs on V_WMMA_F32_16X16X32_F16 (f16 in, f32 accum).
// - Weights pre-packed ONCE per block into LDS f16 in the exact per-lane
//   WMMA-B layout -> B tile load = 32 contiguous bytes/lane (ds_load_b128 x2).
// - One wave = 16 data rows per job, 4 jobs per wave (128 rows/block).
// - LayerNorm / softmax in f32 VALU via LDS staging.
// ---------------------------------------------------------------------------

typedef __attribute__((ext_vector_type(16))) _Float16 v16h;
typedef __attribute__((ext_vector_type(8)))  float    v8f;

#define WAVES   2
#define THREADS (WAVES * 32)
#define JOBS    4                       // 16-row jobs per wave

// packed-weight LDS offsets, in halfs (each tile = 512 halfs = 1KB)
#define OFF_QKV 0                        // 12 tiles (2 layers x 6)
#define OFF_WO  (OFF_QKV + 12 * 512)     //  4 tiles (2 x 2)
#define OFF_W1  (OFF_WO  +  4 * 512)     // 16 tiles (2 x 8)
#define OFF_W2  (OFF_W1  + 16 * 512)     // 16 tiles (2 x (2N x 4K))
#define OFF_HW1 (OFF_W2  + 16 * 512)     // 20 tiles (4N x 5K)
#define WH_HALFS (OFF_HW1 + 20 * 512)    // 34816 halfs = 69632 B

#define ACT_FLOATS_PER_WAVE (512 + 512 + 2048 + 2560)   // X,T,Y,C
#define SMEM_BYTES (WH_HALFS * 2 + WAVES * ACT_FLOATS_PER_WAVE * 4)  // 114688

// ---- weight pre-pack: f32 (N,K) row-major -> f16 tiles in B-lane order ----
// tile (nt,kt) id = nt*Ktiles + kt; within tile, element g = L*16 + e maps to
// W[nt*16 + (L&15)][kt*32 + 8*(L>>4) + (e&7) + 16*(e>>3)]  (lane L holds N=L&15, half picks K)
__device__ inline void pack_mat(const float* __restrict__ W, int Ntiles, int Ktiles,
                                _Float16* __restrict__ dst) {
    const int ldk = Ktiles * 32;
    const int total = Ntiles * Ktiles * 512;
    for (int g = threadIdx.x; g < total; g += THREADS) {
        const int tile = g >> 9;
        const int e512 = g & 511;
        const int L = e512 >> 4, e = e512 & 15;
        const int nt = tile / Ktiles, kt = tile - nt * Ktiles;
        const int n = nt * 16 + (L & 15);
        const int k = kt * 32 + 8 * (L >> 4) + (e & 7) + 16 * (e >> 3);
        dst[g] = (_Float16)W[n * ldk + k];
    }
}

// B tile: one 32B contiguous LDS read per lane (pre-packed layout)
__device__ inline v16h load_b_lds(const _Float16* __restrict__ tile, int lane) {
    return *(const v16h*)(tile + lane * 16);
}

// A tile: 16x32 f16 from f32 LDS (row-major). Lane holds M=lane&15; lane-half
// selects K subset: vgpr v<4 -> K=8h+2v, v>=4 -> K=16+8h+2(v-4).
__device__ inline v16h load_a_lds(const float* __restrict__ buf, int stride, int lane) {
    const int m = lane & 15, half = lane >> 4;
    const float* __restrict__ r = buf + m * stride + 8 * half;
    float lo[8], hi[8];
    *(float4*)(lo)     = *(const float4*)(r);
    *(float4*)(lo + 4) = *(const float4*)(r + 4);
    *(float4*)(hi)     = *(const float4*)(r + 16);
    *(float4*)(hi + 4) = *(const float4*)(r + 20);
    v16h a;
#pragma unroll
    for (int v = 0; v < 4; ++v) {
        a[2 * v]     = (_Float16)lo[2 * v];
        a[2 * v + 1] = (_Float16)lo[2 * v + 1];
    }
#pragma unroll
    for (int v = 0; v < 4; ++v) {
        a[8 + 2 * v]     = (_Float16)hi[2 * v];
        a[8 + 2 * v + 1] = (_Float16)hi[2 * v + 1];
    }
    return a;
}

#define WMMA_F16(a, b, c) \
    __builtin_amdgcn_wmma_f32_16x16x32_f16(false, (a), false, (b), (short)0, (c), false, false)

// C layout: lane holds N = lane&15; vgpr v holds M = v + 8*(lane>>4).
__device__ inline void store_c(float* __restrict__ dst, int stride, int ncol0,
                               const v8f& c, int lane, float bias, bool relu) {
    const int N = lane & 15, half = lane >> 4;
#pragma unroll
    for (int v = 0; v < 8; ++v) {
        float y = c[v] + bias;
        if (relu) y = fmaxf(y, 0.f);
        dst[(v + 8 * half) * stride + ncol0 + N] = y;
    }
}

// Residual add + LayerNorm over D=32, one M-row per lane (lanes 0-15).
__device__ inline void add_ln_row(float* __restrict__ X, const float* __restrict__ R,
                                  int rstride, const float* __restrict__ g,
                                  const float* __restrict__ b, int m) {
    float t[32];
    float mu = 0.f;
#pragma unroll
    for (int j = 0; j < 32; ++j) { t[j] = X[m * 32 + j] + R[m * rstride + j]; mu += t[j]; }
    mu *= (1.f / 32.f);
    float var = 0.f;
#pragma unroll
    for (int j = 0; j < 32; ++j) { const float d = t[j] - mu; var += d * d; }
    var *= (1.f / 32.f);
    const float inv = rsqrtf(var + 1e-5f);
#pragma unroll
    for (int j = 0; j < 32; ++j)
        X[m * 32 + j] = (t[j] - mu) * inv * g[j] + b[j];
}

__global__ void __launch_bounds__(THREADS)
tabtf_fwd_kernel(const long long* __restrict__ cat,      // (B,4) int64
                 const float* __restrict__ cont,          // (B,8)
                 const float* __restrict__ hour_emb,      // (24,32)
                 const float* __restrict__ dow_emb,       // (7,32)
                 const float* __restrict__ month_emb,     // (12,32)
                 const float* __restrict__ grid_emb,      // (10000,32)
                 const float* __restrict__ Wqkv,          // (2,96,32)
                 const float* __restrict__ bqkv,          // (2,96)
                 const float* __restrict__ Wo,            // (2,32,32)
                 const float* __restrict__ bo,            // (2,32)
                 const float* __restrict__ ln1g, const float* __restrict__ ln1b,
                 const float* __restrict__ W1,            // (2,128,32)
                 const float* __restrict__ b1,            // (2,128)
                 const float* __restrict__ W2,            // (2,32,128)
                 const float* __restrict__ b2,            // (2,32)
                 const float* __restrict__ ln2g, const float* __restrict__ ln2b,
                 const float* __restrict__ contW,         // (32,8)
                 const float* __restrict__ contB,         // (32)
                 const float* __restrict__ hW1,           // (64,160)
                 const float* __restrict__ hB1,           // (64)
                 const float* __restrict__ hW2,           // (1,64)
                 const float* __restrict__ hB2,           // (1)
                 float* __restrict__ out, int nrows) {
    extern __shared__ char smem[];
    _Float16* __restrict__ WH = (_Float16*)smem;
    float* __restrict__ act = (float*)(smem + WH_HALFS * 2);

    const int w = threadIdx.x >> 5;
    const int lane = threadIdx.x & 31;

    float* __restrict__ X = act + w * ACT_FLOATS_PER_WAVE;  // 16x32
    float* __restrict__ T = X + 512;                        // 16x32
    float* __restrict__ Y = T + 512;                        // 16x128
    float* __restrict__ C = Y + 2048;                       // 16x160

    // ---- one-time cooperative weight pack into LDS (f16, B-lane order) ----
    pack_mat(Wqkv, 12, 1, WH + OFF_QKV);   // (2*96, 32)
    pack_mat(Wo,    4, 1, WH + OFF_WO);    // (2*32, 32)
    pack_mat(W1,   16, 1, WH + OFF_W1);    // (2*128, 32)
    pack_mat(W2,    4, 4, WH + OFF_W2);    // (2*32, 128)
    pack_mat(hW1,   4, 5, WH + OFF_HW1);   // (64, 160)
    __syncthreads();

    for (int job = 0; job < JOBS; ++job) {
        const int rbase = (blockIdx.x * WAVES * JOBS + w * JOBS + job) * 16;

        // prefetch next job's tokens / cont rows into cache hierarchy
        if (job + 1 < JOBS) {
            const int nr = rbase + 16 + lane;
            if (nr < nrows) {
                __builtin_prefetch(cat + (long long)nr * 4, 0, 1);
                __builtin_prefetch(cont + (long long)nr * 8, 0, 1);
            }
        }

        for (int sr = 0; sr < 4; ++sr) {
            // ---- gather embeddings into X + fill concat h/d/m and cont
            {
                const int m = lane >> 1;              // M-row 0..15
                const int c0 = (lane & 1) * 16;       // column half
                const int rr = m >> 2, t = m & 3;     // sub data-row, token
                int row = rbase + 4 * sr + rr;
                row = row < nrows ? row : nrows - 1;
                const int tok = (int)cat[(long long)row * 4 + t];
                const float* etab = (t == 0) ? hour_emb : (t == 1) ? dow_emb
                                  : (t == 2) ? month_emb : grid_emb;
                const float* e = etab + (long long)tok * 32;
                const int lr = 4 * sr + rr;           // local data row 0..15
#pragma unroll
                for (int j = 0; j < 16; ++j) {
                    const float v = e[c0 + j];
                    X[m * 32 + c0 + j] = v;
                    if (t < 3) C[lr * 160 + 32 * (t + 1) + c0 + j] = v;  // h|d|m
                }
                // cont branch: relu(cont @ contW.T + contB), 4 rows x 32 dims
                const int crr = lane >> 3;            // row 0..3
                const int dg  = (lane & 7) * 4;       // 4 dims each
                int crow = rbase + 4 * sr + crr;
                crow = crow < nrows ? crow : nrows - 1;
                const int clr = 4 * sr + crr;
#pragma unroll
                for (int dd = 0; dd < 4; ++dd) {
                    const int d = dg + dd;
                    float acc = contB[d];
#pragma unroll
                    for (int j = 0; j < 8; ++j)
                        acc += contW[d * 8 + j] * cont[(long long)crow * 8 + j];
                    C[clr * 160 + 128 + d] = fmaxf(acc, 0.f);
                }
            }
            __syncthreads();

            // ---- 2 encoder layers
            for (int li = 0; li < 2; ++li) {
                const float* bqkv_l = bqkv + li * 96;
                const float* bo_l = bo + li * 32;
                const float* b1_l = b1 + li * 128;
                const float* b2_l = b2 + li * 32;
                const _Float16* qkvT = WH + OFF_QKV + li * 6 * 512;
                const _Float16* woT  = WH + OFF_WO  + li * 2 * 512;
                const _Float16* w1T  = WH + OFF_W1  + li * 8 * 512;
                const _Float16* w2T  = WH + OFF_W2  + li * 8 * 512;

                // QKV: (16x32) @ (32x96) -> Y[:, 0..95]
                {
                    const v16h a = load_a_lds(X, 32, lane);
#pragma unroll
                    for (int n = 0; n < 6; ++n) {
                        const v16h b = load_b_lds(qkvT + n * 512, lane);
                        v8f c = {};
                        c = WMMA_F16(a, b, c);
                        store_c(Y, 128, n * 16, c, lane, bqkv_l[n * 16 + (lane & 15)], false);
                    }
                }
                __syncthreads();

                // attention (4 rows x 4 heads x 4 queries = 64 items)
#pragma unroll
                for (int it = 0; it < 2; ++it) {
                    const int idx = lane + 32 * it;
                    const int rr = idx >> 4, hh = (idx >> 2) & 3, q = idx & 3;
                    const float* Q = Y + (rr * 4 + q) * 128 + hh * 8;
                    float sc[4], mx = -1e30f;
#pragma unroll
                    for (int k = 0; k < 4; ++k) {
                        const float* K = Y + (rr * 4 + k) * 128 + 32 + hh * 8;
                        float s = 0.f;
#pragma unroll
                        for (int d = 0; d < 8; ++d) s += Q[d] * K[d];
                        sc[k] = s * 0.35355339059327373f;   // 1/sqrt(8)
                        mx = fmaxf(mx, sc[k]);
                    }
                    float den = 0.f;
#pragma unroll
                    for (int k = 0; k < 4; ++k) { sc[k] = __expf(sc[k] - mx); den += sc[k]; }
                    const float rden = 1.f / den;
#pragma unroll
                    for (int d = 0; d < 8; ++d) {
                        float o = 0.f;
#pragma unroll
                        for (int k = 0; k < 4; ++k)
                            o += sc[k] * Y[(rr * 4 + k) * 128 + 64 + hh * 8 + d];
                        T[(rr * 4 + q) * 32 + hh * 8 + d] = o * rden;
                    }
                }
                __syncthreads();

                // Wo: (16x32) @ (32x32) -> Y[:, 0..31]
                {
                    const v16h a = load_a_lds(T, 32, lane);
#pragma unroll
                    for (int n = 0; n < 2; ++n) {
                        const v16h b = load_b_lds(woT + n * 512, lane);
                        v8f c = {};
                        c = WMMA_F16(a, b, c);
                        store_c(Y, 128, n * 16, c, lane, bo_l[n * 16 + (lane & 15)], false);
                    }
                }
                __syncthreads();

                // x = LN(x + attn)
                if (lane < 16) add_ln_row(X, Y, 128, ln1g + li * 32, ln1b + li * 32, lane);
                __syncthreads();

                // FF1: relu((16x32) @ (32x128)) -> Y
                {
                    const v16h a = load_a_lds(X, 32, lane);
#pragma unroll
                    for (int n = 0; n < 8; ++n) {
                        const v16h b = load_b_lds(w1T + n * 512, lane);
                        v8f c = {};
                        c = WMMA_F16(a, b, c);
                        store_c(Y, 128, n * 16, c, lane, b1_l[n * 16 + (lane & 15)], true);
                    }
                }
                __syncthreads();

                // FF2: (16x128) @ (128x32) -> T
#pragma unroll
                for (int n = 0; n < 2; ++n) {
                    v8f c = {};
#pragma unroll
                    for (int kk = 0; kk < 4; ++kk) {
                        const v16h a = load_a_lds(Y + kk * 32, 128, lane);
                        const v16h b = load_b_lds(w2T + (n * 4 + kk) * 512, lane);
                        c = WMMA_F16(a, b, c);
                    }
                    store_c(T, 32, n * 16, c, lane, b2_l[n * 16 + (lane & 15)], false);
                }
                __syncthreads();

                // x = LN(x + ff)
                if (lane < 16) add_ln_row(X, T, 32, ln2g + li * 32, ln2b + li * 32, lane);
                __syncthreads();
            }

            // ---- z_pool = mean over tokens -> C[:, 0..31]
            {
                const int d = lane;   // 0..31
#pragma unroll
                for (int rr = 0; rr < 4; ++rr) {
                    const int lr = 4 * sr + rr;
                    C[lr * 160 + d] = 0.25f * (X[(rr * 4 + 0) * 32 + d] + X[(rr * 4 + 1) * 32 + d] +
                                               X[(rr * 4 + 2) * 32 + d] + X[(rr * 4 + 3) * 32 + d]);
                }
            }
            __syncthreads();
        }

        // ---- head: relu((16x160) @ (160x64)) -> Y[:, 0..63]
#pragma unroll
        for (int n = 0; n < 4; ++n) {
            v8f c = {};
#pragma unroll
            for (int kk = 0; kk < 5; ++kk) {
                const v16h a = load_a_lds(C + kk * 32, 160, lane);
                const v16h b = load_b_lds(WH + OFF_HW1 + (n * 5 + kk) * 512, lane);
                c = WMMA_F16(a, b, c);
            }
            store_c(Y, 128, n * 16, c, lane, hB1[n * 16 + (lane & 15)], true);
        }
        __syncthreads();

        // ---- logits: (16x64) . hW2 + hB2
        if (lane < 16) {
            const int row = rbase + lane;
            float acc = hB2[0];
#pragma unroll
            for (int j = 0; j < 64; ++j) acc += Y[lane * 128 + j] * hW2[j];
            if (row < nrows) out[row] = acc;
        }
        __syncthreads();   // C/Y reuse in next job
    }
}

extern "C" void kernel_launch(void* const* d_in, const int* in_sizes, int n_in,
                              void* d_out, int out_size, void* d_ws, size_t ws_size,
                              hipStream_t stream) {
    (void)n_in; (void)d_ws; (void)ws_size; (void)out_size;
    const long long* cat  = (const long long*)d_in[0];   // int64 tokens
    const float* cont     = (const float*)d_in[1];
    const float* hour     = (const float*)d_in[2];
    const float* dow      = (const float*)d_in[3];
    const float* month    = (const float*)d_in[4];
    const float* grid     = (const float*)d_in[5];
    const float* Wqkv     = (const float*)d_in[6];
    const float* bqkv     = (const float*)d_in[7];
    const float* Wo       = (const float*)d_in[8];
    const float* bo       = (const float*)d_in[9];
    const float* ln1g     = (const float*)d_in[10];
    const float* ln1b     = (const float*)d_in[11];
    const float* W1       = (const float*)d_in[12];
    const float* b1       = (const float*)d_in[13];
    const float* W2       = (const float*)d_in[14];
    const float* b2       = (const float*)d_in[15];
    const float* ln2g     = (const float*)d_in[16];
    const float* ln2b     = (const float*)d_in[17];
    const float* contW    = (const float*)d_in[18];
    const float* contB    = (const float*)d_in[19];
    const float* hW1      = (const float*)d_in[20];
    const float* hB1      = (const float*)d_in[21];
    const float* hW2      = (const float*)d_in[22];
    const float* hB2      = (const float*)d_in[23];
    float* out = (float*)d_out;

    const int nrows = in_sizes[0] / 4;                     // 262144
    const int rows_per_block = WAVES * JOBS * 16;          // 128
    const int blocks = (nrows + rows_per_block - 1) / rows_per_block;

    tabtf_fwd_kernel<<<dim3(blocks), dim3(THREADS), SMEM_BYTES, stream>>>(
        cat, cont, hour, dow, month, grid, Wqkv, bqkv, Wo, bo, ln1g, ln1b,
        W1, b1, W2, b2, ln2g, ln2b, contW, contB, hW1, hB1, hW2, hB2,
        out, nrows);
}
